// DexterousActor_13529146983064
// MI455X (gfx1250) — compile-verified, matrix operand order checked by reading, not run on hardware
//
#include <hip/hip_runtime.h>
#include <cstdint>
#include <cstddef>

// ---------------- problem constants ----------------
#define B_    32
#define LQ_   96
#define NK_   2048
#define D_    384
#define H_    8
#define DH_   48
#define FF_   1536
#define NL_   6
#define NSELF_ (NK_ + LQ_)   // 2144
#define MQ_   (B_ * LQ_)     // 3072

typedef __bf16 bf16_t;
typedef __attribute__((ext_vector_type(16))) __bf16 v16bf;
typedef __attribute__((ext_vector_type(8)))  __bf16 v8bf;
typedef __attribute__((ext_vector_type(8)))  float  v8f;

// ---------------- CDNA5 async global->LDS staging ----------------
#if defined(__has_builtin)
#  if __has_builtin(__builtin_amdgcn_global_load_async_to_lds_b128)
#    define USE_ASYNC_LDS 1
#  endif
#endif
#ifndef USE_ASYNC_LDS
#  define USE_ASYNC_LDS 0
#endif

#if USE_ASYNC_LDS
// Diagnosed prototype: param0 = 'int __attribute__((vector_size(16))) *'
typedef int v4i_t __attribute__((vector_size(4 * sizeof(int))));
typedef __attribute__((address_space(3))) v4i_t las_v4i;
#endif

// Copy 16B global -> LDS. Async path bypasses VGPRs (ASYNCcnt tracked).
__device__ __forceinline__ void copy16_g2l(bf16_t* lds_dst, const bf16_t* gsrc) {
#if USE_ASYNC_LDS
  __builtin_amdgcn_global_load_async_to_lds_b128((v4i_t*)(void*)gsrc,
                                                 (las_v4i*)lds_dst, 0, 0);
#else
  *(v8bf*)lds_dst = *(const v8bf*)gsrc;
#endif
}
__device__ __forceinline__ void async_join() {
#if USE_ASYNC_LDS
  asm volatile("s_wait_asynccnt 0" ::: "memory");
#endif
}

// ---------------- WMMA helpers ----------------
__device__ __forceinline__ v8f wmma_bf16(v16bf a, v16bf b, v8f c) {
  // D = A(16x32 bf16) x B(32x16 bf16) + C(16x16 f32)
  return __builtin_amdgcn_wmma_f32_16x16x32_bf16(false, a, false, b, (short)0, c,
                                                 false, false);
}

// Build an A/B fragment from a row-major [..][32] tile (caller adds koff).
__device__ __forceinline__ v16bf frag_ld(const bf16_t* p) {
  v8bf lo = *(const v8bf*)p;
  v8bf hi = *(const v8bf*)(p + 16);
  v16bf r;
#pragma unroll
  for (int i = 0; i < 8; ++i) { r[i] = lo[i]; r[i + 8] = hi[i]; }
  return r;
}

__device__ __forceinline__ v8f zero8() {
  v8f z;
#pragma unroll
  for (int i = 0; i < 8; ++i) z[i] = 0.f;
  return z;
}

// ---------------- generic WMMA GEMM (double-buffered LDS, async staging) ----
// C[M,N] = A[M,K](bf16) * W[K,N] + bias, W given transposed as Bt[N,K](bf16).
// EPI: 0 = f32 store, 1 = bf16 store, 2 = bf16 relu store, 3 = bf16 RoPE store.
// chunk!=0 maps A row r -> (r/chunk)*tokpb + base + r%chunk (head slices of x).
template <int EPI>
__global__ __launch_bounds__(256) void gemm_wmma(
    const bf16_t* __restrict__ A, const bf16_t* __restrict__ Bt,
    const float* __restrict__ bias, void* __restrict__ Out,
    int M, int N, int K,
    int chunk, int base, int tokpb,
    int rope_ntok, int rope_thr, int rope_sa, int rope_sb,
    const float* __restrict__ xyz_a, const float* __restrict__ xyz_b) {
  __shared__ __align__(16) bf16_t As[2][128 * 32];
  __shared__ __align__(16) bf16_t Bs[2][128 * 32];

  const int tid  = threadIdx.x;
  const int lane = tid & 31;
  const int wave = tid >> 5;
  const int wm   = wave >> 1;       // 0..3  (M direction, 32 rows each)
  const int wn   = wave & 1;        // 0..1  (N direction, 64 cols each)
  const int lrow = lane & 15;
  const int koff = (lane < 16) ? 0 : 8;
  const int m0 = blockIdx.x * 128;
  const int n0 = blockIdx.y * 128;

  auto stage = [&](int kk, int buf) {
#pragma unroll
    for (int it = 0; it < 2; ++it) {
      int c = tid + it * 256;          // 512 chunks of 8 bf16 (16B) per matrix
      int row = c >> 2, seg = c & 3;
      int gr = m0 + row;
      if (chunk) gr = (gr / chunk) * tokpb + base + (gr % chunk);
      copy16_g2l(&As[buf][row * 32 + seg * 8],
                 A + (size_t)gr * K + kk + seg * 8);
      copy16_g2l(&Bs[buf][row * 32 + seg * 8],
                 Bt + (size_t)(n0 + row) * K + kk + seg * 8);
    }
  };

  v8f acc[2][4];
#pragma unroll
  for (int i = 0; i < 2; ++i)
#pragma unroll
    for (int j = 0; j < 4; ++j) acc[i][j] = zero8();

  const int nk = K >> 5;
  stage(0, 0);
  async_join();
  __syncthreads();

  for (int ki = 0; ki < nk; ++ki) {
    const int buf = ki & 1;
    if (ki + 1 < nk) stage((ki + 1) * 32, buf ^ 1);
    if (ki + 2 < nk) {  // prefetch tile ki+2 toward L2 (global_prefetch_b8)
      if (tid < 128) {
        int gr = m0 + tid;
        if (chunk) gr = (gr / chunk) * tokpb + base + (gr % chunk);
        __builtin_prefetch((const void*)(A + (size_t)gr * K + (ki + 2) * 32), 0, 1);
      } else {
        __builtin_prefetch(
            (const void*)(Bt + (size_t)(n0 + tid - 128) * K + (ki + 2) * 32), 0, 1);
      }
    }

    v16bf af[2], bfr[4];
#pragma unroll
    for (int mf = 0; mf < 2; ++mf)
      af[mf] = frag_ld(&As[buf][(wm * 32 + mf * 16 + lrow) * 32 + koff]);
#pragma unroll
    for (int nf = 0; nf < 4; ++nf)
      bfr[nf] = frag_ld(&Bs[buf][(wn * 64 + nf * 16 + lrow) * 32 + koff]);
#pragma unroll
    for (int mf = 0; mf < 2; ++mf)
#pragma unroll
      for (int nf = 0; nf < 4; ++nf)
        acc[mf][nf] = wmma_bf16(af[mf], bfr[nf], acc[mf][nf]);

    async_join();     // next tile fully resident in LDS
    __syncthreads();  // all waves done reading buf before it is re-staged
  }

  // Epilogue. C layout: lane -> col (lane&15); VGPR r -> row r (lanes<16) / r+8.
#pragma unroll
  for (int mf = 0; mf < 2; ++mf) {
    int rbase = m0 + wm * 32 + mf * 16 + ((lane < 16) ? 0 : 8);
#pragma unroll
    for (int nf = 0; nf < 4; ++nf) {
      int gcol = n0 + wn * 64 + nf * 16 + lrow;
      float bv = bias[gcol];
#pragma unroll
      for (int r = 0; r < 8; ++r) {
        int grow = rbase + r;
        float v = acc[mf][nf][r] + bv;
        if (EPI == 0) {
          ((float*)Out)[(size_t)grow * N + gcol] = v;
        } else if (EPI == 1) {
          ((bf16_t*)Out)[(size_t)grow * N + gcol] = (bf16_t)v;
        } else if (EPI == 2) {
          ((bf16_t*)Out)[(size_t)grow * N + gcol] = (bf16_t)fmaxf(v, 0.f);
        } else {
          // RoPE: adjacent columns live in adjacent lanes -> shuffle pairing.
          int b = grow / rope_ntok, n = grow % rope_ntok;
          const float* xp; int idx;
          if (n < rope_thr) { xp = xyz_a; idx = b * rope_sa + n; }
          else              { xp = xyz_b; idx = b * rope_sb + (n - rope_thr); }
          int axis = gcol >> 7;            // D/3 = 128 per axis
          int j = (gcol >> 1) & 63;        // freq index within axis
          float ang = xp[idx * 3 + axis] * __expf(-0.14391156832f * (float)j);
          float cs = __cosf(ang), sn = __sinf(ang);
          float nb = __shfl_xor(v, 1);
          float o = ((gcol & 1) == 0) ? (v * cs - nb * sn) : (v * cs + nb * sn);
          ((bf16_t*)Out)[(size_t)grow * N + gcol] = (bf16_t)o;
        }
      }
    }
  }
}

// ---------------- attention: scores ----------------
// grid.x = B*H. Per block: Q[96,48] padded to 64 in LDS; each wave owns
// strided 16-col kv tiles, stages K rows into private LDS, 2 WMMAs per tile.
__global__ __launch_bounds__(256) void attn_scores(
    const bf16_t* __restrict__ qp, const bf16_t* __restrict__ kp,
    bf16_t* __restrict__ P, int Nkv) {
  __shared__ __align__(16) bf16_t Qs[96 * 64];
  __shared__ __align__(16) bf16_t Ks[8][16 * 64];
  const int bh = blockIdx.x, b = bh >> 3, h = bh & 7;
  const int tid = threadIdx.x, lane = tid & 31, wave = tid >> 5;
  const int lrow = lane & 15, koff = (lane < 16) ? 0 : 8;

  for (int c = tid; c < 96 * 6; c += 256) {   // 48 bf16 per row = 6 x 16B
    int row = c / 6, seg = c % 6;
    copy16_g2l(&Qs[row * 64 + seg * 8],
               qp + (size_t)(b * LQ_ + row) * D_ + h * DH_ + seg * 8);
  }
  for (int c = tid; c < 96 * 16; c += 256) {  // zero pad cols 48..63
    int row = c >> 4, j = c & 15;
    Qs[row * 64 + 48 + j] = (bf16_t)0.f;
  }
  async_join();
  __syncthreads();

  v16bf af[6][2];
#pragma unroll
  for (int mt = 0; mt < 6; ++mt) {
    af[mt][0] = frag_ld(&Qs[(mt * 16 + lrow) * 64 + koff]);
    af[mt][1] = frag_ld(&Qs[(mt * 16 + lrow) * 64 + 32 + koff]);
  }

  bf16_t* ks = &Ks[wave][0];
  const int NT = Nkv >> 4;
  for (int nt = wave; nt < NT; nt += 8) {
    if (lane < 16) {
      const bf16_t* src = kp + (size_t)(b * Nkv + nt * 16 + lane) * D_ + h * DH_;
#pragma unroll
      for (int seg = 0; seg < 6; ++seg)
        copy16_g2l(&ks[lane * 64 + seg * 8], src + seg * 8);
#pragma unroll
      for (int j = 0; j < 16; ++j) ks[lane * 64 + 48 + j] = (bf16_t)0.f;
    }
    async_join();
    asm volatile("s_wait_dscnt 0" ::: "memory");  // per-wave LDS RAW ordering
    v16bf b0 = frag_ld(&ks[lrow * 64 + koff]);
    v16bf b1 = frag_ld(&ks[lrow * 64 + 32 + koff]);
#pragma unroll
    for (int mt = 0; mt < 6; ++mt) {
      v8f acc = zero8();
      acc = wmma_bf16(af[mt][0], b0, acc);
      acc = wmma_bf16(af[mt][1], b1, acc);
      int qrow0 = mt * 16 + ((lane < 16) ? 0 : 8);
      bf16_t* dst = P + (size_t)(bh * LQ_) * Nkv;
#pragma unroll
      for (int r = 0; r < 8; ++r)
        dst[(size_t)(qrow0 + r) * Nkv + nt * 16 + lrow] =
            (bf16_t)(acc[r] * 0.14433757f);   // 1/sqrt(48)
    }
  }
}

// ---------------- attention: softmax (rows stay in L2) ----------------
__global__ __launch_bounds__(256) void softmax_rows(bf16_t* __restrict__ P, int Nkv) {
  __shared__ float red[8];
  bf16_t* p = P + (size_t)blockIdx.x * (size_t)Nkv;
  const int tid = threadIdx.x;
  float mx = -3.0e38f;
  for (int i = tid; i < Nkv; i += 256) mx = fmaxf(mx, (float)p[i]);
#pragma unroll
  for (int m = 16; m > 0; m >>= 1) mx = fmaxf(mx, __shfl_xor(mx, m));
  if ((tid & 31) == 0) red[tid >> 5] = mx;
  __syncthreads();
  mx = red[0];
#pragma unroll
  for (int w = 1; w < 8; ++w) mx = fmaxf(mx, red[w]);
  __syncthreads();
  float s = 0.f;
  for (int i = tid; i < Nkv; i += 256) {
    float e = __expf((float)p[i] - mx);
    p[i] = (bf16_t)e;
    s += e;
  }
#pragma unroll
  for (int m = 16; m > 0; m >>= 1) s += __shfl_xor(s, m);
  if ((tid & 31) == 0) red[tid >> 5] = s;
  __syncthreads();
  float tot = 0.f;
#pragma unroll
  for (int w = 0; w < 8; ++w) tot += red[w];
  float inv = 1.f / tot;
  for (int i = tid; i < Nkv; i += 256) p[i] = (bf16_t)((float)p[i] * inv);
}

// ---------------- attention: O = P @ V ----------------
// grid.x = B*H*6 (16-row q tiles). A-fragments stream from P (L2 resident);
// V chunks staged transposed per wave in LDS; cross-wave reduce in LDS.
__global__ __launch_bounds__(256) void attn_o(
    const bf16_t* __restrict__ P, const bf16_t* __restrict__ vp,
    bf16_t* __restrict__ O, int Nkv) {
  __shared__ __align__(16) bf16_t Vt[8][48 * 32];  // [dh][k] per wave
  __shared__ float Op[8][16 * 48];
  const int id = blockIdx.x;
  const int mt = id % 6, bh = id / 6, b = bh >> 3, h = bh & 7;
  const int tid = threadIdx.x, lane = tid & 31, wave = tid >> 5;
  const int lrow = lane & 15, koff = (lane < 16) ? 0 : 8;

  v8f acc[3];
#pragma unroll
  for (int nf = 0; nf < 3; ++nf) acc[nf] = zero8();

  bf16_t* vt = &Vt[wave][0];
  const int NC = Nkv >> 5;
  for (int c = wave; c < NC; c += 8) {
    // stage V chunk transposed: lane = kv token within chunk
    const bf16_t* vr = vp + (size_t)(b * Nkv + c * 32 + lane) * D_ + h * DH_;
    v8bf vv[6];
#pragma unroll
    for (int s = 0; s < 6; ++s) vv[s] = *(const v8bf*)(vr + s * 8);
#pragma unroll
    for (int s = 0; s < 6; ++s)
#pragma unroll
      for (int j = 0; j < 8; ++j) vt[(s * 8 + j) * 32 + lane] = vv[s][j];
    asm volatile("s_wait_dscnt 0" ::: "memory");

    const bf16_t* pr =
        P + (size_t)(bh * LQ_ + mt * 16 + lrow) * Nkv + c * 32 + koff;
    v16bf a = frag_ld(pr);
#pragma unroll
    for (int nf = 0; nf < 3; ++nf) {
      v16bf bfr = frag_ld(&vt[(nf * 16 + lrow) * 32 + koff]);
      acc[nf] = wmma_bf16(a, bfr, acc[nf]);
    }
  }
#pragma unroll
  for (int nf = 0; nf < 3; ++nf)
#pragma unroll
    for (int r = 0; r < 8; ++r)
      Op[wave][(r + ((lane < 16) ? 0 : 8)) * 48 + nf * 16 + lrow] = acc[nf][r];
  __syncthreads();
  for (int i = tid; i < 16 * 48; i += 256) {
    int r = i / 48, d = i % 48;
    float s = 0.f;
#pragma unroll
    for (int w = 0; w < 8; ++w) s += Op[w][i];
    O[(size_t)(b * LQ_ + mt * 16 + r) * D_ + h * DH_ + d] = (bf16_t)s;
  }
}

// ---------------- LayerNorm(residual) ----------------
__global__ __launch_bounds__(128) void ln_residual(
    const float* __restrict__ xin, const float* __restrict__ add,
    const float* __restrict__ g, const float* __restrict__ be,
    float* __restrict__ xout, bf16_t* __restrict__ xb) {
  __shared__ float rs[4], rss[4];
  const int row = blockIdx.x, tid = threadIdx.x;
  float v[3];
  float s = 0.f, ss = 0.f;
#pragma unroll
  for (int i = 0; i < 3; ++i) {
    int c = tid + i * 128;
    float t = xin[(size_t)row * D_ + c] + add[(size_t)row * D_ + c];
    v[i] = t; s += t; ss += t * t;
  }
#pragma unroll
  for (int m = 16; m > 0; m >>= 1) { s += __shfl_xor(s, m); ss += __shfl_xor(ss, m); }
  if ((tid & 31) == 0) { rs[tid >> 5] = s; rss[tid >> 5] = ss; }
  __syncthreads();
  s = rs[0] + rs[1] + rs[2] + rs[3];
  ss = rss[0] + rss[1] + rss[2] + rss[3];
  float mean = s * (1.f / D_);
  float var = ss * (1.f / D_) - mean * mean;
  float rstd = rsqrtf(var + 1e-5f);
#pragma unroll
  for (int i = 0; i < 3; ++i) {
    int c = tid + i * 128;
    float o = (v[i] - mean) * rstd * g[c] + be[c];
    xout[(size_t)row * D_ + c] = o;
    xb[(size_t)row * D_ + c] = (bf16_t)o;
  }
}

// ---------------- time embedding -> silu(t_feats) ----------------
__global__ __launch_bounds__(384) void time_embed(
    const float* __restrict__ ts, const float* __restrict__ w1,
    const float* __restrict__ b1, const float* __restrict__ w2,
    const float* __restrict__ b2, float* __restrict__ silu_out) {
  __shared__ float pe[D_], hh[D_];
  const int b = blockIdx.x, d = threadIdx.x;
  float t = ts[b];
  if (d < 192) {
    float fr = __expf(-(9.210340372f / 191.f) * (float)d);
    pe[d] = __sinf(t * fr);
    pe[d + 192] = __cosf(t * fr);
  }
  __syncthreads();
  float a = b1[d];
  for (int k = 0; k < D_; ++k) a += pe[k] * w1[k * D_ + d];
  hh[d] = fmaxf(a, 0.f);
  __syncthreads();
  float o = b2[d];
  for (int k = 0; k < D_; ++k) o += hh[k] * w2[k * D_ + d];
  silu_out[b * D_ + d] = o / (1.f + __expf(-o));
}

// ---------------- AdaLN modulation: mods[l,b,:] = silu_t[b] @ ada_w[l] + ada_b[l]
__global__ __launch_bounds__(128) void compute_mods(
    const float* __restrict__ silu_t, const float* __restrict__ ada_w,
    const float* __restrict__ ada_b, float* __restrict__ mods) {
  __shared__ float st[B_ * D_];
  const int l = blockIdx.x, c = blockIdx.y * 128 + threadIdx.x;
  for (int i = threadIdx.x; i < B_ * D_; i += 128) st[i] = silu_t[i];
  __syncthreads();
  float acc[B_];
  float bb = ada_b[l * 768 + c];
#pragma unroll
  for (int b = 0; b < B_; ++b) acc[b] = bb;
  const float* w = ada_w + (size_t)l * D_ * 768 + c;
  for (int k = 0; k < D_; ++k) {
    float wv = w[(size_t)k * 768];
#pragma unroll
    for (int b = 0; b < B_; ++b) acc[b] += st[b * D_ + k] * wv;
  }
#pragma unroll
  for (int b = 0; b < B_; ++b) mods[((size_t)l * B_ + b) * 768 + c] = acc[b];
}

// ---------------- KV prep: add type embeddings, convert to bf16 ----------------
__global__ __launch_bounds__(256) void prep_kv(
    const float* __restrict__ src, const float* __restrict__ type_emb,
    bf16_t* __restrict__ dst, int Ntok) {
  size_t i = (size_t)blockIdx.x * 256 + threadIdx.x;
  size_t tot = (size_t)B_ * Ntok * D_;
  if (i >= tot) return;
  int col = (int)(i % D_);
  int n = (int)((i / D_) % Ntok);
  float v = src[i];
  if (n == 2046) v += type_emb[3 * D_ + col];
  else if (n == 2047) v += type_emb[4 * D_ + col];
  dst[i] = (bf16_t)v;
}

// ---------------- AdaLN apply on query stream ----------------
__global__ __launch_bounds__(256) void adaln_apply(
    const float* __restrict__ x, const float* __restrict__ mods_l,
    bf16_t* __restrict__ xq) {
  size_t i = (size_t)blockIdx.x * 256 + threadIdx.x;
  if (i >= (size_t)MQ_ * D_) return;
  int col = (int)(i % D_);
  int b = (int)(i / ((size_t)LQ_ * D_));
  float sc = mods_l[b * 768 + col];
  float sh = mods_l[b * 768 + D_ + col];
  xq[i] = (bf16_t)(x[i] * (1.f + sc) + sh);
}

// ---------------- weight transpose + bf16 convert ----------------
__global__ __launch_bounds__(256) void conv_wt(
    const float* __restrict__ W, bf16_t* __restrict__ Wt, int Kd, int Nd, int L) {
  size_t i = (size_t)blockIdx.x * 256 + threadIdx.x;
  size_t tot = (size_t)L * Kd * Nd;
  if (i >= tot) return;
  int n = (int)(i % Nd);
  size_t t = i / Nd;
  int k = (int)(t % Kd);
  int l = (int)(t / Kd);
  Wt[((size_t)l * Nd + n) * Kd + k] = (bf16_t)W[i];
}

// ---------------- final tiny-N head projections ----------------
__global__ __launch_bounds__(256) void head_final(
    const bf16_t* __restrict__ ha, const bf16_t* __restrict__ hq,
    const bf16_t* __restrict__ hb,
    const float* __restrict__ wa, const float* __restrict__ ba,
    const float* __restrict__ wqw, const float* __restrict__ bqv,
    const float* __restrict__ wb, const float* __restrict__ bbv,
    float* __restrict__ out) {
  int i = blockIdx.x * 256 + threadIdx.x;
  if (i >= 1024 * 71) return;
  int r = i / 71, c = i % 71;
  const bf16_t* h; const float* w; const float* bias; int nc, co;
  if (c < 31)      { h = ha; w = wa;  bias = ba;  nc = 31; co = c; }
  else if (c < 62) { h = hq; w = wqw; bias = bqv; nc = 31; co = c - 31; }
  else             { h = hb; w = wb;  bias = bbv; nc = 9;  co = c - 62; }
  float s = bias[co];
  const bf16_t* hr = h + (size_t)r * D_;
  for (int k = 0; k < D_; ++k) s += (float)hr[k] * w[k * nc + co];
  out[i] = s;
}

// =======================================================================
extern "C" void kernel_launch(void* const* d_in, const int* in_sizes, int n_in,
                              void* d_out, int out_size, void* d_ws, size_t ws_size,
                              hipStream_t stream) {
  (void)in_sizes; (void)n_in; (void)out_size; (void)ws_size;
  const float* q        = (const float*)d_in[0];
  const float* k_cross  = (const float*)d_in[1];
  const float* k_self   = (const float*)d_in[2];
  const float* q_xyz    = (const float*)d_in[3];
  const float* k_xyz    = (const float*)d_in[4];
  const float* tsteps   = (const float*)d_in[5];
  const float* t_w1 = (const float*)d_in[6];  const float* t_b1 = (const float*)d_in[7];
  const float* t_w2 = (const float*)d_in[8];  const float* t_b2 = (const float*)d_in[9];
  const float* type_emb = (const float*)d_in[10];
  const float* ada_w = (const float*)d_in[11]; const float* ada_b = (const float*)d_in[12];
  const float* wq = (const float*)d_in[13]; const float* bq = (const float*)d_in[14];
  const float* wk = (const float*)d_in[15]; const float* bk = (const float*)d_in[16];
  const float* wv = (const float*)d_in[17]; const float* bv = (const float*)d_in[18];
  const float* wo = (const float*)d_in[19]; const float* bo = (const float*)d_in[20];
  const float* ln1_g = (const float*)d_in[21]; const float* ln1_b = (const float*)d_in[22];
  const float* fw1 = (const float*)d_in[23]; const float* fb1 = (const float*)d_in[24];
  const float* fw2 = (const float*)d_in[25]; const float* fb2 = (const float*)d_in[26];
  const float* ln2_g = (const float*)d_in[27]; const float* ln2_b = (const float*)d_in[28];
  const float* hw1[3] = {(const float*)d_in[29], (const float*)d_in[35], (const float*)d_in[41]};
  const float* hb1[3] = {(const float*)d_in[30], (const float*)d_in[36], (const float*)d_in[42]};
  const float* hw2[3] = {(const float*)d_in[31], (const float*)d_in[37], (const float*)d_in[43]};
  const float* hb2[3] = {(const float*)d_in[32], (const float*)d_in[38], (const float*)d_in[44]};
  const float* hw3[3] = {(const float*)d_in[33], (const float*)d_in[39], (const float*)d_in[45]};
  const float* hb3[3] = {(const float*)d_in[34], (const float*)d_in[40], (const float*)d_in[46]};

  // ---- workspace bump allocator ----
  uint8_t* wp = (uint8_t*)d_ws;
  auto alloc = [&](size_t bytes) -> void* {
    void* r = wp;
    wp += (bytes + 255) & ~(size_t)255;
    return r;
  };
  float*  silu_t = (float*)alloc((size_t)B_ * D_ * 4);
  float*  mods   = (float*)alloc((size_t)NL_ * B_ * 768 * 4);
  bf16_t* wq_t  = (bf16_t*)alloc((size_t)NL_ * D_ * D_ * 2);
  bf16_t* wk_t  = (bf16_t*)alloc((size_t)NL_ * D_ * D_ * 2);
  bf16_t* wv_t  = (bf16_t*)alloc((size_t)NL_ * D_ * D_ * 2);
  bf16_t* wo_t  = (bf16_t*)alloc((size_t)NL_ * D_ * D_ * 2);
  bf16_t* fw1_t = (bf16_t*)alloc((size_t)NL_ * D_ * FF_ * 2);
  bf16_t* fw2_t = (bf16_t*)alloc((size_t)NL_ * FF_ * D_ * 2);
  bf16_t* hw1t[3]; bf16_t* hw2t[3];
  for (int i = 0; i < 3; ++i) hw1t[i] = (bf16_t*)alloc((size_t)D_ * D_ * 2);
  for (int i = 0; i < 3; ++i) hw2t[i] = (bf16_t*)alloc((size_t)D_ * D_ * 2);
  bf16_t* kc_bf  = (bf16_t*)alloc((size_t)B_ * NK_ * D_ * 2);
  bf16_t* ksf_bf = (bf16_t*)alloc((size_t)B_ * NSELF_ * D_ * 2);
  bf16_t* xq_bf  = (bf16_t*)alloc((size_t)MQ_ * D_ * 2);
  bf16_t* qp     = (bf16_t*)alloc((size_t)MQ_ * D_ * 2);
  bf16_t* kp     = (bf16_t*)alloc((size_t)B_ * NSELF_ * D_ * 2);
  bf16_t* vp     = (bf16_t*)alloc((size_t)B_ * NSELF_ * D_ * 2);
  bf16_t* Pbuf   = (bf16_t*)alloc((size_t)B_ * H_ * LQ_ * NSELF_ * 2);  // L2-resident
  bf16_t* att    = (bf16_t*)alloc((size_t)MQ_ * D_ * 2);
  float*  o32    = (float*)alloc((size_t)MQ_ * D_ * 4);
  float*  x      = (float*)alloc((size_t)MQ_ * D_ * 4);
  bf16_t* x_bf   = (bf16_t*)alloc((size_t)MQ_ * D_ * 2);
  bf16_t* h_ff   = (bf16_t*)alloc((size_t)MQ_ * FF_ * 2);
  bf16_t* h1[3]; bf16_t* h2[3];
  for (int i = 0; i < 3; ++i) h1[i] = (bf16_t*)alloc((size_t)1024 * D_ * 2);
  for (int i = 0; i < 3; ++i) h2[i] = (bf16_t*)alloc((size_t)1024 * D_ * 2);

  auto cdiv = [](size_t a, size_t b) { return (unsigned)((a + b - 1) / b); };

  // ---- one-time prep (re-done every launch; deterministic) ----
  conv_wt<<<cdiv((size_t)NL_ * D_ * D_, 256), 256, 0, stream>>>(wq, wq_t, D_, D_, NL_);
  conv_wt<<<cdiv((size_t)NL_ * D_ * D_, 256), 256, 0, stream>>>(wk, wk_t, D_, D_, NL_);
  conv_wt<<<cdiv((size_t)NL_ * D_ * D_, 256), 256, 0, stream>>>(wv, wv_t, D_, D_, NL_);
  conv_wt<<<cdiv((size_t)NL_ * D_ * D_, 256), 256, 0, stream>>>(wo, wo_t, D_, D_, NL_);
  conv_wt<<<cdiv((size_t)NL_ * D_ * FF_, 256), 256, 0, stream>>>(fw1, fw1_t, D_, FF_, NL_);
  conv_wt<<<cdiv((size_t)NL_ * FF_ * D_, 256), 256, 0, stream>>>(fw2, fw2_t, FF_, D_, NL_);
  for (int i = 0; i < 3; ++i) {
    conv_wt<<<cdiv((size_t)D_ * D_, 256), 256, 0, stream>>>(hw1[i], hw1t[i], D_, D_, 1);
    conv_wt<<<cdiv((size_t)D_ * D_, 256), 256, 0, stream>>>(hw2[i], hw2t[i], D_, D_, 1);
  }
  time_embed<<<B_, 384, 0, stream>>>(tsteps, t_w1, t_b1, t_w2, t_b2, silu_t);
  compute_mods<<<dim3(NL_, 6), 128, 0, stream>>>(silu_t, ada_w, ada_b, mods);
  prep_kv<<<cdiv((size_t)B_ * NK_ * D_, 256), 256, 0, stream>>>(k_cross, type_emb, kc_bf, NK_);
  prep_kv<<<cdiv((size_t)B_ * NSELF_ * D_, 256), 256, 0, stream>>>(k_self, type_emb, ksf_bf, NSELF_);
  (void)hipMemcpyAsync(x, q, (size_t)MQ_ * D_ * 4, hipMemcpyDeviceToDevice, stream);

  // ---- 6 transformer layers ----
  for (int l = 0; l < NL_; ++l) {
    const int Nkv = (l < 2) ? NK_ : NSELF_;
    const bf16_t* kvb = (l < 2) ? kc_bf : ksf_bf;
    const int Mkv = B_ * Nkv;

    adaln_apply<<<cdiv((size_t)MQ_ * D_, 256), 256, 0, stream>>>(
        x, mods + (size_t)l * B_ * 768, xq_bf);

    // Q projection + RoPE (q positions)
    gemm_wmma<3><<<dim3(MQ_ / 128, D_ / 128), 256, 0, stream>>>(
        xq_bf, wq_t + (size_t)l * D_ * D_, bq + l * D_, qp, MQ_, D_, D_,
        0, 0, 0, LQ_, LQ_, LQ_, LQ_, q_xyz, q_xyz);
    // K projection + RoPE (scene positions; self layers append grasp positions)
    gemm_wmma<3><<<dim3(Mkv / 128, D_ / 128), 256, 0, stream>>>(
        kvb, wk_t + (size_t)l * D_ * D_, bk + l * D_, kp, Mkv, D_, D_,
        0, 0, 0, Nkv, NK_, NK_, LQ_, k_xyz, q_xyz);
    // V projection (no rope)
    gemm_wmma<1><<<dim3(Mkv / 128, D_ / 128), 256, 0, stream>>>(
        kvb, wv_t + (size_t)l * D_ * D_, bv + l * D_, vp, Mkv, D_, D_,
        0, 0, 0, 1, 1, 1, 1, nullptr, nullptr);

    attn_scores<<<B_ * H_, 256, 0, stream>>>(qp, kp, Pbuf, Nkv);
    softmax_rows<<<B_ * H_ * LQ_, 256, 0, stream>>>(Pbuf, Nkv);
    attn_o<<<B_ * H_ * 6, 256, 0, stream>>>(Pbuf, vp, att, Nkv);

    gemm_wmma<0><<<dim3(MQ_ / 128, D_ / 128), 256, 0, stream>>>(
        att, wo_t + (size_t)l * D_ * D_, bo + l * D_, o32, MQ_, D_, D_,
        0, 0, 0, 1, 1, 1, 1, nullptr, nullptr);
    ln_residual<<<MQ_, 128, 0, stream>>>(x, o32, ln1_g + l * D_, ln1_b + l * D_, x, x_bf);

    gemm_wmma<2><<<dim3(MQ_ / 128, FF_ / 128), 256, 0, stream>>>(
        x_bf, fw1_t + (size_t)l * D_ * FF_, fb1 + l * FF_, h_ff, MQ_, FF_, D_,
        0, 0, 0, 1, 1, 1, 1, nullptr, nullptr);
    gemm_wmma<0><<<dim3(MQ_ / 128, D_ / 128), 256, 0, stream>>>(
        h_ff, fw2_t + (size_t)l * FF_ * D_, fb2 + l * D_, o32, MQ_, D_, FF_,
        0, 0, 0, 1, 1, 1, 1, nullptr, nullptr);
    ln_residual<<<MQ_, 128, 0, stream>>>(x, o32, ln2_g + l * D_, ln2_b + l * D_, x, x_bf);
  }

  // ---- output heads (act / qf / ob over 32-token slices) ----
  for (int i = 0; i < 3; ++i) {
    gemm_wmma<2><<<dim3(1024 / 128, D_ / 128), 256, 0, stream>>>(
        x_bf, hw1t[i], hb1[i], h1[i], 1024, D_, D_,
        32, i * 32, LQ_, 1, 1, 1, 1, nullptr, nullptr);
    gemm_wmma<2><<<dim3(1024 / 128, D_ / 128), 256, 0, stream>>>(
        h1[i], hw2t[i], hb2[i], h2[i], 1024, D_, D_,
        0, 0, 0, 1, 1, 1, 1, nullptr, nullptr);
  }
  head_final<<<cdiv(1024 * 71, 256), 256, 0, stream>>>(
      h2[0], h2[1], h2[2], hw3[0], hb3[0], hw3[1], hb3[1], hw3[2], hb3[2],
      (float*)d_out);
}
